// UnetUp3_CT_deformable_81080392614518
// MI455X (gfx1250) — matrix operand order repaired and checked by reading, not compiled, not run on hardware
//
#include <hip/hip_runtime.h>
#include <hip/hip_bf16.h>

// ---------------- constants (problem is fixed-size) ----------------
#define DD     24
#define NVOX   13824   // 24^3
#define PVOX   17576   // 26^3
#define CIN    192
#define COUT   64
#define NOFF   81      // 3*27 offset channels
#define NTILES 864     // NVOX/16
#define KTOT1  5184    // 192*27
#define KTOT2  1728    // 64*27

typedef __attribute__((ext_vector_type(16))) __bf16        v16bf;
typedef __attribute__((ext_vector_type(8)))  float         v8f;
typedef __attribute__((ext_vector_type(4)))  unsigned int  v4u;
typedef __attribute__((ext_vector_type(8)))  int           v8i;
typedef __attribute__((ext_vector_type(4)))  int           v4i;

#ifndef __has_builtin
#define __has_builtin(x) 0
#endif
#if __has_builtin(__builtin_amdgcn_tensor_load_to_lds) && \
    __has_builtin(__builtin_amdgcn_s_wait_tensorcnt)
#define HAVE_TDM 1
#if __has_include(<hip/amd_detail/amd_gfx1250_TDM.h>)
#define TDM_6ARG 1
#else
#define TDM_6ARG 0
#endif
#else
#define HAVE_TDM 0
#endif

struct U32x8 { uint4 lo, hi; };

__device__ __forceinline__ float bf2f(unsigned short u) {
  unsigned int x = ((unsigned int)u) << 16;
  return __builtin_bit_cast(float, x);
}
// native RNE f32->bf16 (lets the compiler pick v_cvt_*_bf16 ops)
__device__ __forceinline__ unsigned short f2bf(float f) {
  return __builtin_bit_cast(unsigned short, (__bf16)f);
}
__device__ __forceinline__ v16bf frag_from(uint4 lo, uint4 hi) {
  U32x8 t{lo, hi};
  return __builtin_bit_cast(v16bf, t);
}

#if HAVE_TDM
// TDM: stage a 16-row x 32-col bf16 tile (row stride = row_stride_elems) into LDS.
// 2D D#: tensor/tile = 32 x 16, data_size=2B, dim0 stride = K row pitch.
__device__ __forceinline__ void tdm_stage16x32(const unsigned short* gsrc,
                                               unsigned lds_off,
                                               int row_stride_elems) {
  unsigned long long ga = (unsigned long long)(uintptr_t)gsrc;
  unsigned ga_lo = (unsigned)__builtin_amdgcn_readfirstlane((int)(unsigned)(ga & 0xFFFFFFFFull));
  unsigned ga_hi = (unsigned)__builtin_amdgcn_readfirstlane((int)(unsigned)(ga >> 32));
  unsigned ldu   = (unsigned)__builtin_amdgcn_readfirstlane((int)lds_off);
  v4u g0 = { 1u,                                   // count=1, no gather
             ldu,                                  // lds_addr [63:32]
             ga_lo,                                // global_addr lo
             (ga_hi & 0x1FFFFFFu) | (2u << 30) };  // global_addr hi | type=2
  v8i g1 = { 0x10000,                              // data_size=1 (2 bytes)
             (32 << 16),                           // tensor_dim0 lo16 = 32
             (16 << 16),                           // tensor_dim1 lo16 = 16
             (32 << 16),                           // tile_dim0 = 32
             16,                                   // tile_dim1 = 16
             row_stride_elems,                     // tensor_dim0_stride lo32
             0, 0 };
  v4i gz = { 0, 0, 0, 0 };
#if TDM_6ARG
  v8i z8 = { 0, 0, 0, 0, 0, 0, 0, 0 };
  __builtin_amdgcn_tensor_load_to_lds(g0, g1, gz, gz, z8, 0);
#else
  __builtin_amdgcn_tensor_load_to_lds(g0, g1, gz, gz, 0);
#endif
}
#endif

// ---------------- stage 0: upsample + concat + pad + bf16 channel-last ----
__global__ __launch_bounds__(256)
void k_build_x(const float* __restrict__ in1, const float* __restrict__ in2,
               unsigned short* __restrict__ xcl) {
  int idx = blockIdx.x * 256 + threadIdx.x;
  if (idx >= PVOX * CIN) return;
  int c = idx % CIN;
  int p = idx / CIN;
  int dp = p / 676, r = p % 676, hp = r / 26, wp = r % 26;
  float val = 0.f;
  if (dp >= 1 && dp <= DD && hp >= 1 && hp <= DD && wp >= 1 && wp <= DD) {
    int d = dp - 1, h = hp - 1, w = wp - 1;
    if (c < COUT) {
      val = in1[((c * DD + d) * DD + h) * DD + w];
    } else {
      int ci = c - COUT;
      float cd = d * 0.5f - 0.25f, ch = h * 0.5f - 0.25f, cw = w * 0.5f - 0.25f;
      int id0 = (int)floorf(cd), ih0 = (int)floorf(ch), iw0 = (int)floorf(cw);
      float fd = cd - id0, fh = ch - ih0, fw = cw - iw0;
      int id0c = max(0, min(11, id0)),  id1c = max(0, min(11, id0 + 1));
      int ih0c = max(0, min(11, ih0)),  ih1c = max(0, min(11, ih0 + 1));
      int iw0c = max(0, min(11, iw0)),  iw1c = max(0, min(11, iw0 + 1));
      const float* src = in2 + (size_t)ci * 12 * 12 * 12;
      float s = 0.f;
      #pragma unroll
      for (int k = 0; k < 8; k++) {
        int bd = k & 1, bh = (k >> 1) & 1, bw = (k >> 2) & 1;
        int zd = bd ? id1c : id0c, zh = bh ? ih1c : ih0c, zw = bw ? iw1c : iw0c;
        float wgt = (bd ? fd : 1.f - fd) * (bh ? fh : 1.f - fh) * (bw ? fw : 1.f - fw);
        s += wgt * src[(zd * 12 + zh) * 12 + zw];
      }
      val = s;
    }
  }
  xcl[(size_t)p * CIN + c] = f2bf(val);
}

// ---------------- weight pack: (O,C,27) f32 -> [Mpad][K=t*C+c] bf16 -------
__global__ __launch_bounds__(256)
void k_pack_w(const float* __restrict__ w, unsigned short* __restrict__ wpk,
              int O, int C, int Mpad) {
  int Ktot = C * 27;
  int idx = blockIdx.x * 256 + threadIdx.x;
  if (idx >= Mpad * Ktot) return;
  int m = idx / Ktot, k = idx % Ktot;
  int t = k / C, c = k % C;
  float v = (m < O) ? w[((size_t)m * C + c) * 27 + t] : 0.f;
  wpk[idx] = f2bf(v);
}

// ---------------- implicit-GEMM conv3d (pad=1) via bf16 WMMA --------------
// one wave = 16 out-channels x 16 voxels; K chunk = (tap, 32 channels).
// Weight tile double-buffered in LDS via TDM, overlapped with WMMA.
template<int C, int KTOT>
__global__ __launch_bounds__(128)
void k_conv_gemm(const unsigned short* __restrict__ xcl,
                 const unsigned short* __restrict__ wpk,
                 const float* __restrict__ bias,
                 float* __restrict__ out,
                 int Oreal) {
  constexpr int NCH = KTOT / 32;
#if HAVE_TDM
  __shared__ __align__(16) unsigned short sA[4][2][512];
#endif
  const int lane  = threadIdx.x & 31;
  const int wid   = threadIdx.x >> 5;
  const int tile  = blockIdx.x * 4 + wid;
  const int mtile = tile / NTILES;
  const int ntile = tile - mtile * NTILES;
  const int lhalf = lane >> 4;
  const int l16   = lane & 15;
  const int n = ntile * 16 + l16;
  const int w = n % DD, h = (n / DD) % DD, d = n / (DD * DD);
#if HAVE_TDM
  const unsigned short* wt = wpk + (size_t)(mtile * 16) * KTOT;
  tdm_stage16x32(wt, (unsigned)(uintptr_t)&sA[wid][0][0], KTOT);
#else
  const unsigned short* arow = wpk + (size_t)(mtile * 16 + l16) * KTOT;
#endif
  v8f acc = {0.f, 0.f, 0.f, 0.f, 0.f, 0.f, 0.f, 0.f};
  int i = 0;
  for (int kd = 0; kd < 3; ++kd)
  for (int kh = 0; kh < 3; ++kh)
  for (int kw = 0; kw < 3; ++kw) {
    const unsigned short* brow =
        xcl + (size_t)((d + kd) * 676 + (h + kh) * 26 + (w + kw)) * C + lhalf * 16;
    #pragma unroll
    for (int cb = 0; cb < C; cb += 32) {
#if HAVE_TDM
      int inext = (i + 1 < NCH) ? i + 1 : i;
      tdm_stage16x32(wt + (size_t)inext * 32,
                     (unsigned)(uintptr_t)&sA[wid][(i + 1) & 1][0], KTOT);
      __builtin_amdgcn_s_wait_tensorcnt(1);      // chunk i staged, i+1 in flight
      const uint4* la = (const uint4*)&sA[wid][i & 1][l16 * 32 + lhalf * 16];
      uint4 alo = la[0];
      uint4 ahi = la[1];
#else
      __builtin_prefetch(arow + i * 32 + 128, 0, 3);
      const uint4* ap = (const uint4*)(arow + i * 32 + lhalf * 16);
      uint4 alo = ap[0];
      uint4 ahi = ap[1];
#endif
      const uint4* bp = (const uint4*)(brow + cb);
      uint4 blo = bp[0], bhi = bp[1];
      v16bf av = frag_from(alo, ahi);
      v16bf bv = frag_from(blo, bhi);
      acc = __builtin_amdgcn_wmma_f32_16x16x32_bf16(false, av, false, bv,
                                                    (short)0, acc, false, false);
      ++i;
    }
  }
  #pragma unroll
  for (int r = 0; r < 8; r++) {
    int mm = mtile * 16 + r + lhalf * 8;
    if (mm < Oreal)
      out[(size_t)mm * NVOX + n] = acc[r] + (bias ? bias[mm] : 0.f);
  }
}

// ---------------- deformable conv: fused trilinear gather + WMMA ----------
__global__ __launch_bounds__(128)
void k_deform_gemm(const unsigned short* __restrict__ xcl,
                   const unsigned short* __restrict__ wpk,
                   const float* __restrict__ offs,
                   float* __restrict__ out) {
  constexpr int NCH = KTOT1 / 32;   // 162
#if HAVE_TDM
  __shared__ __align__(16) unsigned short sA[4][2][512];
#endif
  const int lane  = threadIdx.x & 31;
  const int wid   = threadIdx.x >> 5;
  const int tile  = blockIdx.x * 4 + wid;
  const int mtile = tile / NTILES;
  const int ntile = tile - mtile * NTILES;
  const int lhalf = lane >> 4;
  const int l16   = lane & 15;
  const int n = ntile * 16 + l16;
  const int w = n % DD, h = (n / DD) % DD, d = n / (DD * DD);
#if HAVE_TDM
  const unsigned short* wt = wpk + (size_t)(mtile * 16) * KTOT1;
  tdm_stage16x32(wt, (unsigned)(uintptr_t)&sA[wid][0][0], KTOT1);
#else
  const unsigned short* arow = wpk + (size_t)(mtile * 16 + l16) * KTOT1;
#endif
  v8f acc = {0.f, 0.f, 0.f, 0.f, 0.f, 0.f, 0.f, 0.f};
  int i = 0;
  for (int kd = 0; kd < 3; ++kd)
  for (int kh = 0; kh < 3; ++kh)
  for (int kw = 0; kw < 3; ++kw) {
    const int t = (kd * 3 + kh) * 3 + kw;
    // sampling position (per tap, shared by the 6 channel-block WMMAs)
    float od = offs[(size_t)(t)      * NVOX + n];
    float oh = offs[(size_t)(27 + t) * NVOX + n];
    float ow = offs[(size_t)(54 + t) * NVOX + n];
    float pd = fminf(fmaxf((float)(d + kd) + od, 0.f), 25.f);
    float ph = fminf(fmaxf((float)(h + kh) + oh, 0.f), 25.f);
    float pw = fminf(fmaxf((float)(w + kw) + ow, 0.f), 25.f);
    int qd = (int)floorf(pd), qh = (int)floorf(ph), qw = (int)floorf(pw);
    float fd = pd - qd, fh = ph - qh, fw = pw - qw;
    int qd1 = min(qd + 1, 25), qh1 = min(qh + 1, 25), qw1 = min(qw + 1, 25);
    float cwgt[8];
    unsigned crow[8];
    #pragma unroll
    for (int cr = 0; cr < 8; cr++) {
      int bd = cr & 1, bh = (cr >> 1) & 1, bw = (cr >> 2) & 1;
      int zd = bd ? qd1 : qd, zh = bh ? qh1 : qh, zw = bw ? qw1 : qw;
      cwgt[cr] = (bd ? fd : 1.f - fd) * (bh ? fh : 1.f - fh) * (bw ? fw : 1.f - fw);
      crow[cr] = (unsigned)(zd * 676 + zh * 26 + zw) * CIN;
    }
    for (int cb32 = 0; cb32 < CIN; cb32 += 32) {
#if HAVE_TDM
      int inext = (i + 1 < NCH) ? i + 1 : i;
      tdm_stage16x32(wt + (size_t)inext * 32,
                     (unsigned)(uintptr_t)&sA[wid][(i + 1) & 1][0], KTOT1);
      __builtin_amdgcn_s_wait_tensorcnt(1);
      const uint4* la = (const uint4*)&sA[wid][i & 1][l16 * 32 + lhalf * 16];
      uint4 alo = la[0];
      uint4 ahi = la[1];
#else
      const uint4* ap = (const uint4*)(arow + i * 32 + lhalf * 16);
      uint4 alo = ap[0];
      uint4 ahi = ap[1];
#endif
      const int cb = cb32 + lhalf * 16;
      float a16[16];
      #pragma unroll
      for (int j = 0; j < 16; j++) a16[j] = 0.f;
      #pragma unroll
      for (int cr = 0; cr < 8; cr++) {
        const uint4* cp = (const uint4*)(xcl + (size_t)crow[cr] + cb);
        uint4 c0 = cp[0], c1 = cp[1];
        unsigned u[8] = {c0.x, c0.y, c0.z, c0.w, c1.x, c1.y, c1.z, c1.w};
        float wgt = cwgt[cr];
        #pragma unroll
        for (int j = 0; j < 8; j++) {
          a16[2 * j]     += wgt * bf2f((unsigned short)(u[j] & 0xFFFFu));
          a16[2 * j + 1] += wgt * bf2f((unsigned short)(u[j] >> 16));
        }
      }
      // native RNE converts -> compiler can use packed f32->bf16 cvt ops
      v16bf bv;
      #pragma unroll
      for (int j = 0; j < 16; j++) bv[j] = (__bf16)a16[j];
      v16bf av = frag_from(alo, ahi);
      acc = __builtin_amdgcn_wmma_f32_16x16x32_bf16(false, av, false, bv,
                                                    (short)0, acc, false, false);
      ++i;
    }
  }
  #pragma unroll
  for (int r = 0; r < 8; r++) {
    int mm = mtile * 16 + r + lhalf * 8;
    out[(size_t)mm * NVOX + n] = acc[r];
  }
}

// ---------------- BN (batch stats) + ReLU -> padded channel-last bf16 -----
__global__ __launch_bounds__(256)
void k_bn_relu_pack(const float* __restrict__ y,
                    const float* __restrict__ gamma, const float* __restrict__ beta,
                    unsigned short* __restrict__ ycl) {
  const int ch = blockIdx.x;
  const int tid = threadIdx.x;
  __shared__ float ssum[256], ssq[256];
  __shared__ float sscale, sshift;
  float s = 0.f, q = 0.f;
  const float* row = y + (size_t)ch * NVOX;
  for (int i = tid; i < NVOX; i += 256) { float v = row[i]; s += v; q += v * v; }
  ssum[tid] = s; ssq[tid] = q;
  __syncthreads();
  for (int st = 128; st > 0; st >>= 1) {
    if (tid < st) { ssum[tid] += ssum[tid + st]; ssq[tid] += ssq[tid + st]; }
    __syncthreads();
  }
  if (tid == 0) {
    float mean = ssum[0] / (float)NVOX;
    float var  = ssq[0] / (float)NVOX - mean * mean;   // biased, == jnp.var
    float sc = gamma[ch] * rsqrtf(var + 1e-5f);
    sscale = sc; sshift = beta[ch] - mean * sc;
  }
  __syncthreads();
  float sc = sscale, sh = sshift;
  for (int p = tid; p < PVOX; p += 256) {
    int dp = p / 676, r = p % 676, hp = r / 26, wp = r % 26;
    float val = 0.f;
    if (dp >= 1 && dp <= DD && hp >= 1 && hp <= DD && wp >= 1 && wp <= DD) {
      int vi = ((dp - 1) * DD + (hp - 1)) * DD + (wp - 1);
      val = fmaxf(sc * row[vi] + sh, 0.f);
    }
    ycl[(size_t)p * COUT + ch] = f2bf(val);
  }
}

// ---------------- BN (batch stats) + ReLU -> f32 output -------------------
__global__ __launch_bounds__(256)
void k_bn_relu_out(const float* __restrict__ z,
                   const float* __restrict__ gamma, const float* __restrict__ beta,
                   float* __restrict__ outp) {
  const int ch = blockIdx.x;
  const int tid = threadIdx.x;
  __shared__ float ssum[256], ssq[256];
  __shared__ float sscale, sshift;
  float s = 0.f, q = 0.f;
  const float* row = z + (size_t)ch * NVOX;
  for (int i = tid; i < NVOX; i += 256) { float v = row[i]; s += v; q += v * v; }
  ssum[tid] = s; ssq[tid] = q;
  __syncthreads();
  for (int st = 128; st > 0; st >>= 1) {
    if (tid < st) { ssum[tid] += ssum[tid + st]; ssq[tid] += ssq[tid + st]; }
    __syncthreads();
  }
  if (tid == 0) {
    float mean = ssum[0] / (float)NVOX;
    float var  = ssq[0] / (float)NVOX - mean * mean;
    float sc = gamma[ch] * rsqrtf(var + 1e-5f);
    sscale = sc; sshift = beta[ch] - mean * sc;
  }
  __syncthreads();
  float sc = sscale, sh = sshift;
  for (int i = tid; i < NVOX; i += 256) {
    outp[(size_t)ch * NVOX + i] = fmaxf(sc * row[i] + sh, 0.f);
  }
}

// ---------------- host side ----------------
extern "C" void kernel_launch(void* const* d_in, const int* in_sizes, int n_in,
                              void* d_out, int out_size, void* d_ws, size_t ws_size,
                              hipStream_t stream) {
  const float* in1    = (const float*)d_in[0];   // (1,64,24,24,24)
  const float* in2    = (const float*)d_in[1];   // (1,128,12,12,12)
  const float* W_off  = (const float*)d_in[2];   // (81,192,3,3,3)
  const float* b_off  = (const float*)d_in[3];   // (81,)
  const float* W_def  = (const float*)d_in[4];   // (64,192,3,3,3)
  const float* gamma1 = (const float*)d_in[5];
  const float* beta1  = (const float*)d_in[6];
  const float* W2     = (const float*)d_in[7];   // (64,64,3,3,3)
  const float* b2     = (const float*)d_in[8];
  const float* gamma2 = (const float*)d_in[9];
  const float* beta2  = (const float*)d_in[10];
  float* outp = (float*)d_out;

  char* ws = (char*)d_ws;
  size_t off = 0;
  auto carve = [&](size_t bytes) {
    char* p = ws + off;
    off = (off + bytes + 255) & ~(size_t)255;
    return p;
  };
  unsigned short* xcl     = (unsigned short*)carve((size_t)PVOX * CIN * 2);
  unsigned short* wpk_off = (unsigned short*)carve((size_t)96  * KTOT1 * 2);
  unsigned short* wpk_def = (unsigned short*)carve((size_t)64  * KTOT1 * 2);
  unsigned short* wpk_2   = (unsigned short*)carve((size_t)64  * KTOT2 * 2);
  float*          offs    = (float*)carve((size_t)NOFF * NVOX * 4);
  float*          ybuf    = (float*)carve((size_t)COUT * NVOX * 4);
  unsigned short* ycl     = (unsigned short*)carve((size_t)PVOX * COUT * 2);
  float*          zbuf    = (float*)carve((size_t)COUT * NVOX * 4);
  (void)ws_size; (void)n_in; (void)in_sizes; (void)out_size;

  // 0) x = concat(in1, upsample(in2)) -> padded channel-last bf16
  {
    int total = PVOX * CIN;
    k_build_x<<<(total + 255) / 256, 256, 0, stream>>>(in1, in2, xcl);
  }
  // 0b) pack weights to bf16 [M][t*C+c]
  {
    int t1 = 96 * KTOT1;
    k_pack_w<<<(t1 + 255) / 256, 256, 0, stream>>>(W_off, wpk_off, NOFF, CIN, 96);
    int t2 = 64 * KTOT1;
    k_pack_w<<<(t2 + 255) / 256, 256, 0, stream>>>(W_def, wpk_def, COUT, CIN, 64);
    int t3 = 64 * KTOT2;
    k_pack_w<<<(t3 + 255) / 256, 256, 0, stream>>>(W2, wpk_2, COUT, COUT, 64);
  }
  // 1) offsets = conv3d(x, W_off) + b_off   [M=81(pad 96), K=5184, N=13824]
  k_conv_gemm<CIN, KTOT1><<<(6 * NTILES) / 4, 128, 0, stream>>>(
      xcl, wpk_off, b_off, offs, NOFF);
  // 2) y = deform_conv3d(x, offsets, W_def) [M=64, K=5184, N=13824]
  k_deform_gemm<<<(4 * NTILES) / 4, 128, 0, stream>>>(xcl, wpk_def, offs, ybuf);
  // 3) BN1 + ReLU -> padded channel-last bf16
  k_bn_relu_pack<<<COUT, 256, 0, stream>>>(ybuf, gamma1, beta1, ycl);
  // 4) z = conv3d(y, W2) + b2               [M=64, K=1728, N=13824]
  k_conv_gemm<COUT, KTOT2><<<(4 * NTILES) / 4, 128, 0, stream>>>(
      ycl, wpk_2, b2, zbuf, COUT);
  // 5) BN2 + ReLU -> d_out (f32)
  k_bn_relu_out<<<COUT, 256, 0, stream>>>(zbuf, gamma2, beta2, outp);
}
